// MultiHeadSelfAttention_11261404250490
// MI455X (gfx1250) — matrix operand order repaired and checked by reading, not compiled, not run on hardware
//
#include <hip/hip_runtime.h>
#include <hip/hip_bf16.h>

typedef __bf16 bf16_t;
typedef __attribute__((ext_vector_type(8)))  __bf16 v8bf;
typedef __attribute__((ext_vector_type(16))) __bf16 v16bf;
typedef __attribute__((ext_vector_type(8)))  float  v8f;
typedef __attribute__((ext_vector_type(4)))  float  v4f;

#define DMODEL 1024
#define SEQ    2048
#define BATCH  2
#define NHEAD  16
#define DKH    64

__device__ __forceinline__ v8f wmma_bf16(v16bf a, v16bf b, v8f c) {
  // D = A(16x32 bf16) * B(32x16 bf16) + C(16x16 f32)
  return __builtin_amdgcn_wmma_f32_16x16x32_bf16(false, a, false, b, (short)0, c, false, false);
}

// Load a 16-element bf16 vector as two contiguous 16-byte chunks.
__device__ __forceinline__ v16bf ld_pair(const bf16_t* p0, const bf16_t* p1) {
  v8bf a = *(const v8bf*)p0;
  v8bf b = *(const v8bf*)p1;
  v16bf r;
#pragma unroll
  for (int i = 0; i < 8; ++i) { r[i] = a[i]; r[i + 8] = b[i]; }
  return r;
}

// Vectorized f32 -> bf16 (8 elements per thread; n must be a multiple of 2048).
__global__ __launch_bounds__(256) void f32_to_bf16_k(const float* __restrict__ in,
                                                     bf16_t* __restrict__ out, int n) {
  int i = (blockIdx.x * 256 + threadIdx.x) * 8;
  if (i < n) {
    v4f x0 = *(const v4f*)(in + i);
    v4f x1 = *(const v4f*)(in + i + 4);
    v8bf y;
#pragma unroll
    for (int j = 0; j < 4; ++j) { y[j] = (bf16_t)x0[j]; y[j + 4] = (bf16_t)x1[j]; }
    *(v8bf*)(out + i) = y;
  }
}

// C[M,N] = alpha * A[M,K] @ Bw[N,K]^T   (A, Bw row-major bf16, K contiguous)
// mode 0: bf16 out, row-major [M][N]
// mode 1: bf16 out, "V-transposed": out[b][n][s] with m = b*SEQ + s  (N must be DMODEL)
// mode 2: f32 out, row-major [M][N]
// One wave computes a 16x64 tile. 2-stage register double-buffered K-loop with
// UNCONDITIONAL prefetch (epilogue peeled) so LOADcnt is statically known and the
// compiler can leave the prefetch clause in flight across the WMMAs.
__global__ __launch_bounds__(256) void gemm_bf16_k(
    const bf16_t* __restrict__ A, const bf16_t* __restrict__ Bw,
    void* __restrict__ Out, int M, int N, int K, int mode, float alpha) {
  int lane = threadIdx.x & 31;
  int hi   = lane >> 4;   // 0 or 1
  int lw   = lane & 15;
  int wid  = blockIdx.x * 8 + (threadIdx.x >> 5);
  int ngrp = N >> 6;
  int mtile = wid / ngrp;
  int ng    = wid - mtile * ngrp;
  int m0 = mtile << 4;
  int n0 = ng << 6;

  const bf16_t* arow  = A  + (size_t)(m0 + lw) * K;
  const bf16_t* brow0 = Bw + (size_t)(n0 + lw) * K;

  auto ldA = [&](int kk) -> v16bf {
    // A 16x32: per lane contiguous k = kk + 16*hi .. +15
    return ld_pair(arow + kk + hi * 16, arow + kk + hi * 16 + 8);
  };
  auto ldB = [&](int t, int kk) -> v16bf {
    // B 32x16: per lane contiguous k = kk + 16*hi .. +15
    const bf16_t* bp = brow0 + (size_t)(t * 16) * K + kk + hi * 16;
    return ld_pair(bp, bp + 8);
  };

  // Two pipeline stages in registers (k and k+32).
  v16bf a0  = ldA(0);
  v16bf b00 = ldB(0, 0), b01 = ldB(1, 0), b02 = ldB(2, 0), b03 = ldB(3, 0);
  v16bf a1  = ldA(32);
  v16bf b10 = ldB(0, 32), b11 = ldB(1, 32), b12 = ldB(2, 32), b13 = ldB(3, 32);

  v8f acc[4] = {};
  // Main loop: prefetches are unconditional and provably in bounds
  // (kk+64 <= K-64, kk+96 <= K-32), so no branches around VMEM clauses.
  for (int kk = 0; kk < K - 64; kk += 64) {
    acc[0] = wmma_bf16(a0, b00, acc[0]);
    acc[1] = wmma_bf16(a0, b01, acc[1]);
    acc[2] = wmma_bf16(a0, b02, acc[2]);
    acc[3] = wmma_bf16(a0, b03, acc[3]);
    a0  = ldA(kk + 64);
    b00 = ldB(0, kk + 64); b01 = ldB(1, kk + 64);
    b02 = ldB(2, kk + 64); b03 = ldB(3, kk + 64);
    acc[0] = wmma_bf16(a1, b10, acc[0]);
    acc[1] = wmma_bf16(a1, b11, acc[1]);
    acc[2] = wmma_bf16(a1, b12, acc[2]);
    acc[3] = wmma_bf16(a1, b13, acc[3]);
    a1  = ldA(kk + 96);
    b10 = ldB(0, kk + 96); b11 = ldB(1, kk + 96);
    b12 = ldB(2, kk + 96); b13 = ldB(3, kk + 96);
  }
  // Epilogue: last 64 of K, no prefetch.
  acc[0] = wmma_bf16(a0, b00, acc[0]);
  acc[1] = wmma_bf16(a0, b01, acc[1]);
  acc[2] = wmma_bf16(a0, b02, acc[2]);
  acc[3] = wmma_bf16(a0, b03, acc[3]);
  acc[0] = wmma_bf16(a1, b10, acc[0]);
  acc[1] = wmma_bf16(a1, b11, acc[1]);
  acc[2] = wmma_bf16(a1, b12, acc[2]);
  acc[3] = wmma_bf16(a1, b13, acc[3]);

#pragma unroll
  for (int t = 0; t < 4; ++t) {
#pragma unroll
    for (int r = 0; r < 8; ++r) {
      int m = m0 + r + hi * 8;          // C layout: row = r + 8*(lane>=16)
      int n = n0 + t * 16 + lw;         // col = lane&15
      float v = acc[t][r] * alpha;
      if (mode == 0) {
        ((bf16_t*)Out)[(size_t)m * N + n] = (bf16_t)v;
      } else if (mode == 1) {
        int bb = m >> 11;               // m = b*SEQ + s
        int s  = m & (SEQ - 1);
        ((bf16_t*)Out)[((size_t)bb * N + n) * SEQ + s] = (bf16_t)v;
      } else {
        ((float*)Out)[(size_t)m * N + n] = v;
      }
    }
  }
}

// Flash attention: one wave per (b, h, 16-query tile). Q pre-scaled by 1/sqrt(dk).
// Q,K: bf16 [b][s][h*64+d] row-major.  Vt: bf16 [b][h*64+d][s].  Attn out: bf16 [b][s][h*64+d].
// Pipelined: V loads issued before score WMMAs; next step's K tiles prefetched
// UNCONDITIONALLY (address clamped in bounds) so the load counter stays static.
__global__ __launch_bounds__(256) void attn_fwd_k(
    const bf16_t* __restrict__ Q, const bf16_t* __restrict__ Km,
    const bf16_t* __restrict__ Vt, bf16_t* __restrict__ Attn) {
  __shared__ __align__(16) bf16_t pbuf[8][16 * 32];   // per-wave P staging (C->A relayout)

  int lane = threadIdx.x & 31;
  int hi = lane >> 4, lw = lane & 15;
  int wave = threadIdx.x >> 5;
  int wid = blockIdx.x * 8 + wave;
  int qt = wid & (SEQ / 16 - 1);
  int h  = (wid >> 7) & (NHEAD - 1);
  int b  = wid >> 11;
  int qbase = qt << 4;

  const bf16_t* qrow = Q + (size_t)(b * SEQ + qbase + lw) * DMODEL + h * DKH;
  v16bf aq0 = ld_pair(qrow + 0  + hi * 16, qrow + 8  + hi * 16);
  v16bf aq1 = ld_pair(qrow + 32 + hi * 16, qrow + 40 + hi * 16);

  const bf16_t* vbase = Vt + ((size_t)b * DMODEL + h * DKH) * SEQ;
  bf16_t* pb = &pbuf[wave][0];

  auto ldK = [&](int kb, int t, int ks) -> v16bf {
    const bf16_t* p = Km + (size_t)(b * SEQ + kb + t * 16 + lw) * DMODEL + h * DKH
                         + ks * 32 + hi * 16;
    return ld_pair(p, p + 8);
  };
  auto ldV = [&](int kb, int t) -> v16bf {
    const bf16_t* p = vbase + (size_t)(t * 16 + lw) * SEQ + kb + hi * 16;
    return ld_pair(p, p + 8);
  };

  v8f o[4] = {};
  float mrow[8], lrow[8];
#pragma unroll
  for (int r = 0; r < 8; ++r) { mrow[r] = -3.0e38f; lrow[r] = 0.f; }

  int kend = qbase + 16;

  // Preload K tiles for the first key step.
  v16bf k00 = ldK(0, 0, 0), k01 = ldK(0, 0, 1);
  v16bf k10 = ldK(0, 1, 0), k11 = ldK(0, 1, 1);

  for (int kb = 0; kb < kend; kb += 32) {
    // V tiles for this step: issued early, consumed after softmax.
    v16bf v0 = ldV(kb, 0), v1 = ldV(kb, 1), v2 = ldV(kb, 2), v3 = ldV(kb, 3);

    // ---- scores for 32 keys: two 16x16 C tiles ----
    v8f c0 = {}, c1 = {};
    c0 = wmma_bf16(aq0, k00, c0);
    c1 = wmma_bf16(aq0, k10, c1);
    c0 = wmma_bf16(aq1, k01, c0);
    c1 = wmma_bf16(aq1, k11, c1);

    // Unconditional prefetch of next step's K tiles; clamp keeps addresses valid
    // (last step refetches step 0, data unused). Keeps LOADcnt statically known.
    int kbn = (kb + 32 < kend) ? (kb + 32) : 0;
    k00 = ldK(kbn, 0, 0); k01 = ldK(kbn, 0, 1);
    k10 = ldK(kbn, 1, 0); k11 = ldK(kbn, 1, 1);

    // ---- causal mask (wave-uniform branch; per-lane cndmask inside) ----
    if (kb + 31 > qbase) {
#pragma unroll
      for (int r = 0; r < 8; ++r) {
        int q = qbase + r + hi * 8;
        c0[r] = (kb + lw      > q) ? -1.0e30f : c0[r];
        c1[r] = (kb + 16 + lw > q) ? -1.0e30f : c1[r];
      }
    }

    // ---- online softmax (row = r + 8*hi, reduce across the 16-lane half) ----
    float scl[8];
    v8f p0, p1;
#pragma unroll
    for (int r = 0; r < 8; ++r) {
      float t = fmaxf(c0[r], c1[r]);
      t = fmaxf(t, __shfl_xor(t, 1, 32));
      t = fmaxf(t, __shfl_xor(t, 2, 32));
      t = fmaxf(t, __shfl_xor(t, 4, 32));
      t = fmaxf(t, __shfl_xor(t, 8, 32));
      float mn = fmaxf(mrow[r], t);
      scl[r] = __expf(mrow[r] - mn);
      p0[r] = __expf(c0[r] - mn);
      p1[r] = __expf(c1[r] - mn);
      float rs = p0[r] + p1[r];
      rs += __shfl_xor(rs, 1, 32);
      rs += __shfl_xor(rs, 2, 32);
      rs += __shfl_xor(rs, 4, 32);
      rs += __shfl_xor(rs, 8, 32);
      lrow[r] = lrow[r] * scl[r] + rs;
      mrow[r] = mn;
    }
#pragma unroll
    for (int t = 0; t < 4; ++t)
#pragma unroll
      for (int r = 0; r < 8; ++r) o[t][r] *= scl[r];

    // ---- P (C layout) -> LDS row-major [16][32] -> A layout (in-wave, in-order DS) ----
#pragma unroll
    for (int r = 0; r < 8; ++r) {
      int m = r + hi * 8;
      pb[m * 32 + lw]      = (bf16_t)p0[r];
      pb[m * 32 + 16 + lw] = (bf16_t)p1[r];
    }
    v8bf x0 = *(const v8bf*)(pb + lw * 32 + hi * 16);
    v8bf x1 = *(const v8bf*)(pb + lw * 32 + 8 + hi * 16);
    v16bf ap;
#pragma unroll
    for (int i = 0; i < 8; ++i) { ap[i] = x0[i]; ap[i + 8] = x1[i]; }

    // ---- O += P @ Vt : k = 32 keys, 4 dk tiles of 16 ----
    o[0] = wmma_bf16(ap, v0, o[0]);
    o[1] = wmma_bf16(ap, v1, o[1]);
    o[2] = wmma_bf16(ap, v2, o[2]);
    o[3] = wmma_bf16(ap, v3, o[3]);
  }

  // ---- normalize and store attention output (bf16, [b][s][h*64+d]) ----
#pragma unroll
  for (int t = 0; t < 4; ++t) {
#pragma unroll
    for (int r = 0; r < 8; ++r) {
      int q = qbase + r + hi * 8;
      float v = o[t][r] / lrow[r];
      Attn[(size_t)(b * SEQ + q) * DMODEL + h * DKH + t * 16 + lw] = (bf16_t)v;
    }
  }
}

extern "C" void kernel_launch(void* const* d_in, const int* in_sizes, int n_in,
                              void* d_out, int out_size, void* d_ws, size_t ws_size,
                              hipStream_t stream) {
  const float* x  = (const float*)d_in[0];
  const float* wq = (const float*)d_in[1];
  const float* wk = (const float*)d_in[2];
  const float* wv = (const float*)d_in[3];
  const float* wo = (const float*)d_in[4];

  char* ws = (char*)d_ws;
  const size_t XB = (size_t)BATCH * SEQ * DMODEL * sizeof(bf16_t);   // 8 MiB
  const size_t WB = (size_t)DMODEL * DMODEL * sizeof(bf16_t);        // 2 MiB
  bf16_t* Xbf = (bf16_t*)(ws);
  bf16_t* Wqb = (bf16_t*)(ws + XB);
  bf16_t* Wkb = (bf16_t*)(ws + XB + WB);
  bf16_t* Wvb = (bf16_t*)(ws + XB + 2 * WB);
  bf16_t* Wob = (bf16_t*)(ws + XB + 3 * WB);
  bf16_t* Qb  = (bf16_t*)(ws + XB + 4 * WB);
  bf16_t* Kb  = (bf16_t*)(ws + 2 * XB + 4 * WB);
  bf16_t* Vt  = (bf16_t*)(ws + 3 * XB + 4 * WB);
  bf16_t* Attn = Xbf;   // Xbf is dead after the QKV projections; reuse for attn output

  const int nx = BATCH * SEQ * DMODEL;
  const int nw = DMODEL * DMODEL;
  f32_to_bf16_k<<<nx / 2048, 256, 0, stream>>>(x,  Xbf, nx);
  f32_to_bf16_k<<<nw / 2048, 256, 0, stream>>>(wq, Wqb, nw);
  f32_to_bf16_k<<<nw / 2048, 256, 0, stream>>>(wk, Wkb, nw);
  f32_to_bf16_k<<<nw / 2048, 256, 0, stream>>>(wv, Wvb, nw);
  f32_to_bf16_k<<<nw / 2048, 256, 0, stream>>>(wo, Wob, nw);

  const int M = BATCH * SEQ;                           // 4096
  const int nblk = (M / 16) * (DMODEL / 64) / 8;       // 512 blocks x 8 waves
  gemm_bf16_k<<<nblk, 256, 0, stream>>>(Xbf, Wqb, Qb, M, DMODEL, DMODEL, 0, 0.125f);
  gemm_bf16_k<<<nblk, 256, 0, stream>>>(Xbf, Wkb, Kb, M, DMODEL, DMODEL, 0, 1.0f);
  gemm_bf16_k<<<nblk, 256, 0, stream>>>(Xbf, Wvb, Vt, M, DMODEL, DMODEL, 1, 1.0f);

  const int ablk = (BATCH * NHEAD * (SEQ / 16)) / 8;   // 512 blocks x 8 waves
  attn_fwd_k<<<ablk, 256, 0, stream>>>(Qb, Kb, Vt, Attn);

  gemm_bf16_k<<<nblk, 256, 0, stream>>>(Attn, Wob, d_out, M, DMODEL, DMODEL, 2, 1.0f);
}
